// Decoder_68616397521502
// MI455X (gfx1250) — matrix operand order repaired and checked by reading, not compiled, not run on hardware
//
#include <hip/hip_runtime.h>

// ---------------- problem dimensions ----------------
#define H_DIM   1024
#define B_DIM   128
#define F_DIM   16
#define S_DIM   256
#define V_DIM   64
#define FB_DIM  (F_DIM * B_DIM)       // 2048
#define SB_DIM  (S_DIM * B_DIM)       // 32768

typedef __attribute__((ext_vector_type(16))) __bf16 bf16x16;
typedef __attribute__((ext_vector_type(8)))  float  f32x8;

#define ACT_RELU 1
#define ACT_TANH 2
#define EP_STORE 0
#define EP_DOT   1

template <int ACT>
__device__ __forceinline__ float act_apply(float v) {
  if (ACT == ACT_RELU) return v > 0.f ? v : 0.f;
  if (ACT == ACT_TANH) return tanhf(v);
  return v;
}

// Load 16 fp32 (two 32B-aligned chunks of 8) and convert to a bf16 WMMA fragment.
__device__ __forceinline__ bf16x16 load_frag16(const float* __restrict__ p0,
                                               const float* __restrict__ p1) {
  float t[16];
  *(float4*)(t + 0)  = *(const float4*)(p0 + 0);
  *(float4*)(t + 4)  = *(const float4*)(p0 + 4);
  *(float4*)(t + 8)  = *(const float4*)(p1 + 0);
  *(float4*)(t + 12) = *(const float4*)(p1 + 4);
  bf16x16 r;
#pragma unroll
  for (int j = 0; j < 16; ++j) r[j] = (__bf16)t[j];
  return r;
}

// ---------------------------------------------------------------------------
// Generic WMMA GEMM:  out = act( [A0|A1|A2] @ W^T + bias )
//   - A segments row-major [*, H_DIM]; per-segment row = (m & mask)
//     (mask = rows-1; implements jnp.broadcast_to of [B,H] operands).
//   - W row-major [H_DIM, NSEG*H_DIM] (column block s = segment s).
//   - EP_STORE: out[M, H_DIM] = act(C).
//   - EP_DOT:   scores[m] += sum_n act(C[m,n]) * sw[n]  (fused e@sw reduction,
//     16-lane shuffle reduce + atomicAdd; avoids materializing e).
// Wave tile 32Mx64N: 2 A-frags x 4 B-frags -> 8 WMMA per k-step (B reused
// across both M rows). Block = 8 waves as 4Mx2N -> 128x128 block tile.
// NSEG is a template parameter; the segment loop fully unrolls so every
// address chains directly to a kernel argument (global_load, not flat_load).
// k-loop unrolled x2 so the scheduler has two independent fragment sets to
// rename across (hides the 4-slot WMMA->VALU WAR hazard on B registers).
// A-frag layout (16-bit A 16x32): lanes0-15 K={0-7,16-23}, lanes16-31 +8.
// B-frag layout (16-bit B 32x16): lanes0-15 K={0-7,16-23}, lanes16-31 +8.
// C layout: VGPR i, lanes0-15 -> M=i, lanes16-31 -> M=i+8; N = lane&15.
// ---------------------------------------------------------------------------
template <int ACT, int MODE, int NSEG>
__global__ __launch_bounds__(256)
void k_gemm(const float* __restrict__ A0, unsigned m0,
            const float* __restrict__ A1, unsigned m1,
            const float* __restrict__ A2, unsigned m2,
            const float* __restrict__ W,
            const float* __restrict__ bias,
            float* __restrict__ out,
            const float* __restrict__ sw) {
  constexpr int KTOT = NSEG * H_DIM;
  const int lane  = threadIdx.x & 31;
  const int wave  = threadIdx.x >> 5;
  const int tileM = blockIdx.y * 128 + (wave & 3) * 32;
  const int tileN = blockIdx.x * 128 + (wave >> 2) * 64;
  const int r16   = lane & 15;
  const int hi    = lane >> 4;   // 0 or 1
  const int ka    = hi * 8;      // A-fragment K base
  const int kb    = hi * 8;      // B-fragment K base

  f32x8 zero8;
#pragma unroll
  for (int j = 0; j < 8; ++j) zero8[j] = 0.f;
  f32x8 acc[2][4];
#pragma unroll
  for (int mi = 0; mi < 2; ++mi)
#pragma unroll
    for (int nt = 0; nt < 4; ++nt) acc[mi][nt] = zero8;

#pragma unroll
  for (int s = 0; s < NSEG; ++s) {
    const float* __restrict__ As = (s == 0) ? A0 : (s == 1) ? A1 : A2;
    const unsigned ms            = (s == 0) ? m0 : (s == 1) ? m1 : m2;
    // hoisted row base pointers; k-loop uses immediate offsets off these
    const float* __restrict__ ar0 =
        As + (size_t)((unsigned)(tileM + r16) & ms) * H_DIM + ka;
    const float* __restrict__ ar1 =
        As + (size_t)((unsigned)(tileM + 16 + r16) & ms) * H_DIM + ka;
    const float* __restrict__ wr[4];
#pragma unroll
    for (int nt = 0; nt < 4; ++nt)
      wr[nt] = W + s * H_DIM + (size_t)(tileN + nt * 16 + r16) * KTOT + kb;
#pragma unroll 2
    for (int k = 0; k < H_DIM; k += 32) {
      bf16x16 af0 = load_frag16(ar0 + k, ar0 + k + 16);
      bf16x16 af1 = load_frag16(ar1 + k, ar1 + k + 16);
#pragma unroll
      for (int nt = 0; nt < 4; ++nt) {
        bf16x16 bfrag = load_frag16(wr[nt] + k, wr[nt] + k + 16);
        acc[0][nt] = __builtin_amdgcn_wmma_f32_16x16x32_bf16(
            false, af0, false, bfrag, (short)0, acc[0][nt], false, false);
        acc[1][nt] = __builtin_amdgcn_wmma_f32_16x16x32_bf16(
            false, af1, false, bfrag, (short)0, acc[1][nt], false, false);
      }
    }
  }

  if (MODE == EP_STORE) {
#pragma unroll
    for (int mi = 0; mi < 2; ++mi) {
      const int mbase = tileM + mi * 16 + hi * 8;
#pragma unroll
      for (int nt = 0; nt < 4; ++nt) {
        const int n  = tileN + nt * 16 + r16;
        const float bv = bias[n];
#pragma unroll
        for (int i = 0; i < 8; ++i)
          out[(size_t)(mbase + i) * H_DIM + n] =
              act_apply<ACT>(acc[mi][nt][i] + bv);
      }
    }
  } else {
    float srow[2][8];
#pragma unroll
    for (int mi = 0; mi < 2; ++mi)
#pragma unroll
      for (int i = 0; i < 8; ++i) srow[mi][i] = 0.f;
#pragma unroll
    for (int nt = 0; nt < 4; ++nt) {
      const int n  = tileN + nt * 16 + r16;
      const float bv = bias[n];
      const float wv = sw[n];
#pragma unroll
      for (int mi = 0; mi < 2; ++mi)
#pragma unroll
        for (int i = 0; i < 8; ++i)
          srow[mi][i] += act_apply<ACT>(acc[mi][nt][i] + bv) * wv;
    }
#pragma unroll
    for (int off = 1; off < 16; off <<= 1) {
#pragma unroll
      for (int mi = 0; mi < 2; ++mi)
#pragma unroll
        for (int i = 0; i < 8; ++i)
          srow[mi][i] += __shfl_xor(srow[mi][i], off, 16);
    }
    if (r16 == 0) {
#pragma unroll
      for (int mi = 0; mi < 2; ++mi) {
        const int mbase = tileM + mi * 16 + hi * 8;
#pragma unroll
        for (int i = 0; i < 8; ++i)
          atomicAdd(&out[mbase + i], srow[mi][i]);
      }
    }
  }
}

// ---------------- auxiliary kernels ----------------

__global__ void k_zero(float* __restrict__ p, int n) {
  int i = blockIdx.x * 256 + threadIdx.x;
  if (i < n) p[i] = 0.f;
}

// last_embedded[b,:] = word_dict_vec[b, ids[b], :]
__global__ void k_gather(const int* __restrict__ ids,
                         const float* __restrict__ wdv,
                         float* __restrict__ out) {
  const int b  = blockIdx.x;
  const int id = ids[b];
  const float* src = wdv + ((size_t)b * V_DIM + id) * H_DIM;
  for (int h = threadIdx.x; h < H_DIM; h += 256)
    out[(size_t)b * H_DIM + h] = src[h];
}

// y = (x - mean) * rsqrt(var + 1e-5) * g + b   (per row of H_DIM)
__global__ void k_layernorm(const float* __restrict__ x,
                            const float* __restrict__ g,
                            const float* __restrict__ bb,
                            float* __restrict__ y) {
  __shared__ float red[256];
  const int row = blockIdx.x, t = threadIdx.x;
  const float* xr = x + (size_t)row * H_DIM;
  float s = 0.f;
  for (int h = t; h < H_DIM; h += 256) s += xr[h];
  red[t] = s; __syncthreads();
  for (int o = 128; o > 0; o >>= 1) { if (t < o) red[t] += red[t + o]; __syncthreads(); }
  const float mean = red[0] * (1.f / H_DIM);
  __syncthreads();
  float v = 0.f;
  for (int h = t; h < H_DIM; h += 256) { float d = xr[h] - mean; v += d * d; }
  red[t] = v; __syncthreads();
  for (int o = 128; o > 0; o >>= 1) { if (t < o) red[t] += red[t + o]; __syncthreads(); }
  const float inv = rsqrtf(red[0] * (1.f / H_DIM) + 1e-5f);
  for (int h = t; h < H_DIM; h += 256)
    y[(size_t)row * H_DIM + h] = (xr[h] - mean) * inv * g[h] + bb[h];
}

// f -= c * z  (elementwise)
__global__ void k_fmcz(float* __restrict__ f, const float* __restrict__ c,
                       const float* __restrict__ z, int n) {
  int i = blockIdx.x * 256 + threadIdx.x;
  if (i < n) f[i] -= c[i] * z[i];
}

// fh = (1-sigmoid(score)) * feed_hidden + sigmoid(score) * feedback  (per row)
__global__ void k_fed_gate(const float* __restrict__ scores,
                           const float* __restrict__ fh_in,
                           const float* __restrict__ fb,
                           float* __restrict__ fh_out) {
  const int r = blockIdx.x;
  const float g = 1.f / (1.f + __expf(-scores[r]));
  const size_t base = (size_t)r * H_DIM;
  for (int h = threadIdx.x; h < H_DIM; h += 256)
    fh_out[base + h] = (1.f - g) * fh_in[base + h] + g * fb[base + h];
}

// per b: softmax over F of scores[f*B+b]; new_hidden[b,:] = sum_f w_f * values[f*B+b,:]
__global__ void k_sel_combine(const float* __restrict__ scores,
                              const float* __restrict__ values,
                              float* __restrict__ nh) {
  const int b = blockIdx.x;
  float sc[F_DIM];
  float m = -1e30f;
#pragma unroll
  for (int f = 0; f < F_DIM; ++f) { sc[f] = scores[f * B_DIM + b]; m = fmaxf(m, sc[f]); }
  float sum = 0.f;
#pragma unroll
  for (int f = 0; f < F_DIM; ++f) { sc[f] = __expf(sc[f] - m); sum += sc[f]; }
  const float inv = 1.f / sum;
  for (int h = threadIdx.x; h < H_DIM; h += 256) {
    float acc = 0.f;
#pragma unroll
    for (int f = 0; f < F_DIM; ++f)
      acc += sc[f] * values[((size_t)f * B_DIM + b) * H_DIM + h];
    nh[(size_t)b * H_DIM + h] = acc * inv;
  }
}

// per b: mask + softmax over S of scores[s*B+b]; context[b,:] = sum_s w_s * enc[s*B+b,:]
__global__ void k_attn_combine(const float* __restrict__ scores,
                               const unsigned char* __restrict__ mask,
                               const float* __restrict__ enc,
                               float* __restrict__ ctx) {
  __shared__ float red[256];
  __shared__ float w[256];
  const int b = blockIdx.x, t = threadIdx.x;   // blockDim == S_DIM == 256
  float s = scores[t * B_DIM + b];
  if (mask[b * S_DIM + t]) s = -1e12f;
  red[t] = s; __syncthreads();
  for (int o = 128; o > 0; o >>= 1) { if (t < o) red[t] = fmaxf(red[t], red[t + o]); __syncthreads(); }
  const float m = red[0]; __syncthreads();
  const float e = __expf(s - m);
  red[t] = e; __syncthreads();
  for (int o = 128; o > 0; o >>= 1) { if (t < o) red[t] += red[t + o]; __syncthreads(); }
  w[t] = e / red[0]; __syncthreads();
  for (int h = t; h < H_DIM; h += 256) {
    float acc = 0.f;
    for (int si = 0; si < S_DIM; ++si)
      acc += w[si] * enc[((size_t)si * B_DIM + b) * H_DIM + h];
    ctx[(size_t)b * H_DIM + h] = acc;
  }
}

// ---------------- launcher ----------------

extern "C" void kernel_launch(void* const* d_in, const int* in_sizes, int n_in,
                              void* d_out, int out_size, void* d_ws, size_t ws_size,
                              hipStream_t stream) {
  const int*           ids   = (const int*)d_in[0];
  const float*         fhid  = (const float*)d_in[1];   // feed_hidden [F,B,H]
  const float*         outp  = (const float*)d_in[2];   // output [B,H]
  const float*         wdv   = (const float*)d_in[3];   // word_dict_vec [B,V,H]
  const float*         enc   = (const float*)d_in[4];   // encoder_outputs [S,B,H]
  const unsigned char* smask = (const unsigned char*)d_in[5];
  const float* gf_iw=(const float*)d_in[6],  *gf_ib=(const float*)d_in[7];
  const float* gf_cw=(const float*)d_in[8],  *gf_cb=(const float*)d_in[9];
  const float* gf_zw=(const float*)d_in[10], *gf_zb=(const float*)d_in[11];
  const float* gf_ng=(const float*)d_in[12], *gf_nb=(const float*)d_in[13];
  const float* gf_og=(const float*)d_in[14], *gf_ob=(const float*)d_in[15];
  const float* gr_iw=(const float*)d_in[16], *gr_ib=(const float*)d_in[17];
  const float* gr_og=(const float*)d_in[24], *gr_ob=(const float*)d_in[25];
  const float* gs_iw=(const float*)d_in[26], *gs_ib=(const float*)d_in[27];
  const float* gs_cw=(const float*)d_in[28], *gs_cb=(const float*)d_in[29];
  const float* gs_zw=(const float*)d_in[30], *gs_zb=(const float*)d_in[31];
  const float* gs_ng=(const float*)d_in[32], *gs_nb=(const float*)d_in[33];
  const float* gs_og=(const float*)d_in[34], *gs_ob=(const float*)d_in[35];
  const float* gc_iw=(const float*)d_in[36], *gc_ib=(const float*)d_in[37];
  const float* gc_cw=(const float*)d_in[38], *gc_cb=(const float*)d_in[39];
  const float* gc_zw=(const float*)d_in[40], *gc_zb=(const float*)d_in[41];
  const float* gc_ng=(const float*)d_in[42], *gc_nb=(const float*)d_in[43];
  const float* gc_og=(const float*)d_in[44], *gc_ob=(const float*)d_in[45];
  const float* attn_w=(const float*)d_in[46], *attn_b=(const float*)d_in[47], *attn_sw=(const float*)d_in[48];
  const float* sel_w =(const float*)d_in[49], *sel_b =(const float*)d_in[50], *sel_sw =(const float*)d_in[51];
  const float* fed_w =(const float*)d_in[52], *fed_b =(const float*)d_in[53], *fed_sw =(const float*)d_in[54];

  const size_t BH  = (size_t)B_DIM * H_DIM;       // 131072
  const size_t FBH = (size_t)FB_DIM * H_DIM;      // 2097152

  // outputs: [out0 (B,H)] [fh (F,B,H)] [new_hidden (B,H)]
  float* out0 = (float*)d_out;
  float* FH   = out0 + BH;
  float* NH   = FH + FBH;

  // workspace layout (~54 MB of f32)
  float* ws = (float*)d_ws;
  float* LAST  = ws;               // B*H
  float* SOL   = LAST + BH;        // B*H
  float* FBUF  = SOL + BH;         // FB*H
  float* TBUF  = FBUF + FBH;
  float* CBUF  = TBUF + FBH;
  float* ZBUF  = CBUF + FBH;
  float* FEEDB = ZBUF + FBH;
  float* VALS  = FEEDB + FBH;
  float* SCF   = VALS + FBH;       // FB
  float* SCS   = SCF + FB_DIM;     // FB
  float* SCA   = SCS + FB_DIM;     // S*B
  float* CTX   = SCA + SB_DIM;     // B*H
  float* G1    = CTX + BH;         // 4x B*H small gc buffers
  float* G2    = G1 + BH;
  float* G3    = G2 + BH;
  float* G4    = G3 + BH;

  const dim3 blk(256);
  const dim3 gB(H_DIM / 128, B_DIM / 128);    // M = 128
  const dim3 gFB(H_DIM / 128, FB_DIM / 128);  // M = 2048
  const dim3 gSB(H_DIM / 128, SB_DIM / 128);  // M = 32768

  // zero the three atomic score buffers (contiguous)
  {
    int n = 2 * FB_DIM + SB_DIM;
    k_zero<<<(n + 255) / 256, blk, 0, stream>>>(SCF, n);
  }

  // 1) last_embedded gather
  k_gather<<<B_DIM, blk, 0, stream>>>(ids, wdv, LAST);

  // 2) gr (solution): f = relu([fh[:1]|output|last_emb] @ gr_iw^T + ib); ln
  k_gemm<ACT_RELU, EP_STORE, 3><<<gB, blk, 0, stream>>>(
      fhid, B_DIM - 1, outp, B_DIM - 1, LAST, B_DIM - 1,
      gr_iw, gr_ib, G1, nullptr);
  k_layernorm<<<B_DIM, blk, 0, stream>>>(G1, gr_og, gr_ob, SOL);

  // 3) gf (feedback): inputs = [feed_hidden, broadcast(solution)]
  k_gemm<ACT_RELU, EP_STORE, 2><<<gFB, blk, 0, stream>>>(
      fhid, FB_DIM - 1, SOL, B_DIM - 1, nullptr, 0,
      gf_iw, gf_ib, FBUF, nullptr);
  for (int l = 0; l < 2; ++l) {
    k_layernorm<<<FB_DIM, blk, 0, stream>>>(FBUF, gf_ng + l * H_DIM, gf_nb + l * H_DIM, TBUF);
    k_gemm<ACT_RELU, EP_STORE, 2><<<gFB, blk, 0, stream>>>(
        fhid, FB_DIM - 1, SOL, B_DIM - 1, nullptr, 0,
        gf_cw + (size_t)l * H_DIM * 2 * H_DIM, gf_cb + l * H_DIM, CBUF, nullptr);
    k_gemm<ACT_RELU, EP_STORE, 3><<<gFB, blk, 0, stream>>>(
        CBUF, FB_DIM - 1, TBUF, FB_DIM - 1, fhid, FB_DIM - 1,
        gf_zw + (size_t)l * H_DIM * 3 * H_DIM, gf_zb + l * H_DIM, ZBUF, nullptr);
    k_fmcz<<<(int)(FBH / 256), blk, 0, stream>>>(FBUF, CBUF, ZBUF, (int)FBH);
  }
  k_layernorm<<<FB_DIM, blk, 0, stream>>>(FBUF, gf_og, gf_ob, FEEDB);

  // 4) UpdateGate: scores = relu([sol|feed_hidden]@fed_w^T + b) . fed_sw ; blend
  k_gemm<ACT_RELU, EP_DOT, 2><<<gFB, blk, 0, stream>>>(
      SOL, B_DIM - 1, fhid, FB_DIM - 1, nullptr, 0,
      fed_w, fed_b, SCF, fed_sw);
  k_fed_gate<<<FB_DIM, blk, 0, stream>>>(SCF, fhid, FEEDB, FH);

  // 5) gs (values): inputs = [broadcast(solution), fh]
  k_gemm<ACT_RELU, EP_STORE, 2><<<gFB, blk, 0, stream>>>(
      SOL, B_DIM - 1, FH, FB_DIM - 1, nullptr, 0,
      gs_iw, gs_ib, FBUF, nullptr);
  for (int l = 0; l < 2; ++l) {
    k_layernorm<<<FB_DIM, blk, 0, stream>>>(FBUF, gs_ng + l * H_DIM, gs_nb + l * H_DIM, TBUF);
    k_gemm<ACT_RELU, EP_STORE, 2><<<gFB, blk, 0, stream>>>(
        SOL, B_DIM - 1, FH, FB_DIM - 1, nullptr, 0,
        gs_cw + (size_t)l * H_DIM * 2 * H_DIM, gs_cb + l * H_DIM, CBUF, nullptr);
    k_gemm<ACT_RELU, EP_STORE, 3><<<gFB, blk, 0, stream>>>(
        CBUF, FB_DIM - 1, TBUF, FB_DIM - 1, SOL, B_DIM - 1,
        gs_zw + (size_t)l * H_DIM * 3 * H_DIM, gs_zb + l * H_DIM, ZBUF, nullptr);
    k_fmcz<<<(int)(FBH / 256), blk, 0, stream>>>(FBUF, CBUF, ZBUF, (int)FBH);
  }
  k_layernorm<<<FB_DIM, blk, 0, stream>>>(FBUF, gs_og, gs_ob, VALS);

  // 6) sel attention over F -> new_hidden
  k_gemm<ACT_TANH, EP_DOT, 2><<<gFB, blk, 0, stream>>>(
      SOL, B_DIM - 1, FH, FB_DIM - 1, nullptr, 0,
      sel_w, sel_b, SCS, sel_sw);
  k_sel_combine<<<B_DIM, blk, 0, stream>>>(SCS, VALS, NH);

  // 7) encoder attention (big fused GEMM+dot) -> context
  k_gemm<ACT_TANH, EP_DOT, 2><<<gSB, blk, 0, stream>>>(
      NH, B_DIM - 1, enc, SB_DIM - 1, nullptr, 0,
      attn_w, attn_b, SCA, attn_sw);
  k_attn_combine<<<B_DIM, blk, 0, stream>>>(SCA, smask, enc, CTX);

  // 8) gc: minus_gate([new_hidden, context]) -> out0
  k_gemm<ACT_RELU, EP_STORE, 2><<<gB, blk, 0, stream>>>(
      NH, B_DIM - 1, CTX, B_DIM - 1, nullptr, 0,
      gc_iw, gc_ib, G1, nullptr);
  for (int l = 0; l < 2; ++l) {
    k_layernorm<<<B_DIM, blk, 0, stream>>>(G1, gc_ng + l * H_DIM, gc_nb + l * H_DIM, G2);
    k_gemm<ACT_RELU, EP_STORE, 2><<<gB, blk, 0, stream>>>(
        NH, B_DIM - 1, CTX, B_DIM - 1, nullptr, 0,
        gc_cw + (size_t)l * H_DIM * 2 * H_DIM, gc_cb + l * H_DIM, G3, nullptr);
    k_gemm<ACT_RELU, EP_STORE, 3><<<gB, blk, 0, stream>>>(
        G3, B_DIM - 1, G2, B_DIM - 1, NH, B_DIM - 1,
        gc_zw + (size_t)l * H_DIM * 3 * H_DIM, gc_zb + l * H_DIM, G4, nullptr);
    k_fmcz<<<(int)(BH / 256), blk, 0, stream>>>(G1, G3, G4, (int)BH);
  }
  k_layernorm<<<B_DIM, blk, 0, stream>>>(G1, gc_og, gc_ob, out0);
}
